// Model_21388937134679
// MI455X (gfx1250) — compile-verified, matrix operand order checked by reading, not run on hardware
//
#include <hip/hip_runtime.h>
#include <math.h>

// ---------------------------------------------------------------------------
// Problem constants (from reference): S=4 species, A=32768 atoms/species,
// F=512 input features, H=256 hidden, 512 structures.
// ---------------------------------------------------------------------------
#define SPECIES 4
#define NATOMS  32768
#define NFEAT   512
#define NHID    256
#define NSTRUCT 512
#define TILE_M  64      // atoms per workgroup
#define KT1     (NFEAT / 32)   // 16 k-tiles for layer 1
#define KT2     (NHID  / 32)   // 8  k-tiles for layers 2/3
#define NT      (NHID  / 16)   // 16 n-tiles per layer

// bf16-weight workspace layout (uint32 units)
#define WS_OFF1 0u                                   // W1: S*16nt*16kt*32lane*8u = 262144
#define WS_OFF2 (WS_OFF1 + SPECIES * NT * KT1 * 32u * 8u)   // 262144
#define WS_OFF3 (WS_OFF2 + SPECIES * NT * KT2 * 32u * 8u)   // 393216
#define WS_END  (WS_OFF3 + SPECIES * NT * KT2 * 32u * 8u)   // 524288 uints = 2 MB

typedef __attribute__((ext_vector_type(16))) __bf16 v16bf;
typedef __attribute__((ext_vector_type(8)))  float  v8f;

// ---------------------------------------------------------------------------
// helpers: f32 <-> bf16 (round-to-nearest-even)
// ---------------------------------------------------------------------------
__device__ __forceinline__ unsigned short f2bf(float x) {
  unsigned u = __float_as_uint(x);
  u += 0x7FFFu + ((u >> 16) & 1u);
  return (unsigned short)(u >> 16);
}
__device__ __forceinline__ unsigned pack2(float a, float b) {
  return (unsigned)f2bf(a) | ((unsigned)f2bf(b) << 16);
}
__device__ __forceinline__ float bf2f(unsigned short h) {
  return __uint_as_float(((unsigned)h) << 16);
}

// A/B fragment K-offset for VGPR-pair slot v (0..7), lane-half kh (0/1).
// Matches the ISA 16-bit 16x32 A-matrix layout (CDNA5 ISA, 7.12.2):
// VGPR0-3: K 0..7 (lanes 0-15) / 8..15 (lanes 16-31); VGPR4-7: +16.
__device__ __forceinline__ int frag_koff(int v, int kh) {
  return ((v >> 2) << 4) + (kh << 3) + ((v & 3) << 1);
}

// ---------------------------------------------------------------------------
// kernel: out[j] = energy_shift  (atomics accumulate on top)
// ---------------------------------------------------------------------------
__global__ void init_out_kernel(float* __restrict__ out,
                                const float* __restrict__ shift) {
  int i = blockIdx.x * blockDim.x + threadIdx.x;
  if (i < NSTRUCT) out[i] = shift[0];
}

// ---------------------------------------------------------------------------
// kernel: one-time f32 -> bf16 weight conversion into WMMA B-fragment layout.
// One thread per (fragment, lane): writes 16 bf16 (32 contiguous bytes).
// 2048 fragments * 32 lanes = 65536 threads.
// ---------------------------------------------------------------------------
__global__ void prep_weights_kernel(const float* __restrict__ W1,
                                    const float* __restrict__ W2,
                                    const float* __restrict__ W3,
                                    unsigned* __restrict__ ws) {
  int gid  = blockIdx.x * blockDim.x + threadIdx.x;
  int lane = gid & 31;
  int frag = gid >> 5;          // 0..2047
  if (frag >= SPECIES * 512) return;
  int s  = frag >> 9;           // species
  int fr = frag & 511;          // 0..255: W1, 256..383: W2, 384..511: W3

  const float* src; unsigned dst; int nt, kt;
  if (fr < 256) {
    nt = fr >> 4; kt = fr & 15;
    src = W1 + (size_t)s * NFEAT * NHID;
    dst = WS_OFF1 + (((unsigned)(s * NT + nt) * KT1 + kt) * 32u + lane) * 8u;
  } else if (fr < 384) {
    int f2 = fr - 256; nt = f2 >> 3; kt = f2 & 7;
    src = W2 + (size_t)s * NHID * NHID;
    dst = WS_OFF2 + (((unsigned)(s * NT + nt) * KT2 + kt) * 32u + lane) * 8u;
  } else {
    int f2 = fr - 384; nt = f2 >> 3; kt = f2 & 7;
    src = W3 + (size_t)s * NHID * NHID;
    dst = WS_OFF3 + (((unsigned)(s * NT + nt) * KT2 + kt) * 32u + lane) * 8u;
  }

  int n  = nt * 16 + (lane & 15);
  int kh = lane >> 4;
  unsigned tmp[8];
#pragma unroll
  for (int v = 0; v < 8; ++v) {
    int k = kt * 32 + frag_koff(v, kh);
    float a = src[(size_t)k * NHID + n];
    float b = src[(size_t)(k + 1) * NHID + n];
    tmp[v] = pack2(a, b);
  }
  ((uint4*)(ws + dst))[0] = make_uint4(tmp[0], tmp[1], tmp[2], tmp[3]);
  ((uint4*)(ws + dst))[1] = make_uint4(tmp[4], tmp[5], tmp[6], tmp[7]);
}

// ---------------------------------------------------------------------------
// one fused GEMM layer: D = silu(A @ W + b), A fragments in LDS, W fragments
// in global (L2-resident), D written bf16 row-major [64][256] into LDS.
// Each of the 8 waves computes 8 of the 64 (mt,nt) 16x16 tiles.
// ---------------------------------------------------------------------------
template <int KT>
__device__ __forceinline__ void gemm_layer(const unsigned* __restrict__ afrag,
                                           const unsigned* __restrict__ wfrag,
                                           const float* __restrict__ bias,
                                           unsigned* __restrict__ rowbuf,
                                           int wave, int lane) {
  unsigned short* rb = (unsigned short*)rowbuf;
#pragma unroll
  for (int j = 0; j < 8; ++j) {
    int t  = wave * 8 + j;
    int mt = t >> 4;            // 0..3
    int nt = t & 15;            // 0..15
    v8f acc = {};
#pragma unroll
    for (int kt = 0; kt < KT; ++kt) {
      v16bf a = *(const v16bf*)(afrag + ((mt * KT + kt) * 32 + lane) * 8);
      v16bf b = *(const v16bf*)(wfrag + ((nt * KT + kt) * 32 + lane) * 8);
      acc = __builtin_amdgcn_wmma_f32_16x16x32_bf16(
          false, a, false, b, (short)0, acc, false, false);
    }
    // D layout: VGPR r, lanes 0-15 -> M=r, N=lane; lanes 16-31 -> M=8+r.
    int col   = nt * 16 + (lane & 15);
    int rbase = mt * 16 + ((lane >> 4) << 3);
    float bv  = bias[col];
#pragma unroll
    for (int r = 0; r < 8; ++r) {
      float x = acc[r] + bv;
      x = x / (1.0f + __expf(-x));          // silu
      rb[(rbase + r) * NHID + col] = f2bf(x);
    }
  }
}

// re-stage bf16 row-major [64][256] (rowbuf) into A-fragment layout (frag)
// for the next layer (KT2 = 8 k-tiles). Adjacent K pairs share one uint.
__device__ __forceinline__ void restage(const unsigned* __restrict__ rowbuf,
                                        unsigned* __restrict__ frag, int tid) {
#pragma unroll
  for (int i = 0; i < 4; ++i) {
    int slot = tid + i * 256;   // 0..1023
    int lane = slot & 31;
    int fidx = slot >> 5;       // 0..31
    int mt = fidx >> 3, kt = fidx & 7;
    int m  = mt * 16 + (lane & 15);
    int kh = lane >> 4;
    unsigned tmp[8];
#pragma unroll
    for (int v = 0; v < 8; ++v) {
      int k = kt * 32 + frag_koff(v, kh);
      tmp[v] = rowbuf[m * (NHID / 2) + (k >> 1)];
    }
    unsigned* dst = frag + (fidx * 32 + lane) * 8;
    ((uint4*)dst)[0] = make_uint4(tmp[0], tmp[1], tmp[2], tmp[3]);
    ((uint4*)dst)[1] = make_uint4(tmp[4], tmp[5], tmp[6], tmp[7]);
  }
}

// ---------------------------------------------------------------------------
// main fused kernel: block = 256 threads (8 wave32), 64 atoms of one species.
// LDS (dynamic, 96 KB): frag[16384 u32] (64 KB) + rowbuf[8192 u32] (32 KB).
// ---------------------------------------------------------------------------
__global__ void mlp_kernel(const float* __restrict__ feat,
                           const int*   __restrict__ sid,
                           const float* __restrict__ b1,
                           const float* __restrict__ b2,
                           const float* __restrict__ b3,
                           const float* __restrict__ W4,
                           const float* __restrict__ b4,
                           const unsigned* __restrict__ ws,
                           float* __restrict__ out) {
  extern __shared__ unsigned smem[];
  unsigned* frag   = smem;           // 16384 uints
  unsigned* rowbuf = smem + 16384;   // 8192 uints

  int tid  = threadIdx.x;
  int lane = tid & 31;
  int wave = tid >> 5;
  int s    = blockIdx.y;
  int a0   = blockIdx.x * TILE_M;

  // ---- stage features (f32 global -> bf16 A-fragments in LDS) ----
  const float* xbase = feat + ((size_t)s * NATOMS + a0) * NFEAT;
#pragma unroll
  for (int i = 0; i < 8; ++i) {
    int slot = tid + i * 256;   // 0..2047  (64 frags * 32 lanes)
    int l    = slot & 31;
    int fidx = slot >> 5;       // 0..63
    int mt = fidx >> 4, kt = fidx & 15;
    int m  = mt * 16 + (l & 15);
    int kh = l >> 4;
    const float* rowp = xbase + (size_t)m * NFEAT + kt * 32;
    unsigned tmp[8];
#pragma unroll
    for (int v = 0; v < 8; ++v) {
      float2 val = *(const float2*)(rowp + frag_koff(v, kh));
      tmp[v] = pack2(val.x, val.y);
    }
    unsigned* dst = frag + (fidx * 32 + l) * 8;
    ((uint4*)dst)[0] = make_uint4(tmp[0], tmp[1], tmp[2], tmp[3]);
    ((uint4*)dst)[1] = make_uint4(tmp[4], tmp[5], tmp[6], tmp[7]);
  }
  __syncthreads();

  // ---- layer 1: [64,512] @ [512,256] ----
  gemm_layer<KT1>(frag, ws + WS_OFF1 + (unsigned)s * NT * KT1 * 32u * 8u,
                  b1 + s * NHID, rowbuf, wave, lane);
  __syncthreads();
  restage(rowbuf, frag, tid);
  __syncthreads();

  // ---- layer 2: [64,256] @ [256,256] ----
  gemm_layer<KT2>(frag, ws + WS_OFF2 + (unsigned)s * NT * KT2 * 32u * 8u,
                  b2 + s * NHID, rowbuf, wave, lane);
  __syncthreads();
  restage(rowbuf, frag, tid);
  __syncthreads();

  // ---- layer 3: [64,256] @ [256,256] ----
  gemm_layer<KT2>(frag, ws + WS_OFF3 + (unsigned)s * NT * KT2 * 32u * 8u,
                  b3 + s * NHID, rowbuf, wave, lane);
  __syncthreads();

  // ---- layer 4: e = h3 @ W4 + b4, then scatter-add per structure ----
  {
    int r = tid & 63;           // atom within tile
    int q = tid >> 6;           // quarter of H
    const unsigned short* rb = (const unsigned short*)rowbuf;
    const float* w4s = W4 + s * NHID;
    float p = 0.f;
#pragma unroll 8
    for (int c = q * 64; c < q * 64 + 64; ++c)
      p += bf2f(rb[r * NHID + c]) * w4s[c];
    float* red = (float*)frag;  // frag region reused as f32 scratch
    red[q * 64 + r] = p;
    __syncthreads();
    if (tid < TILE_M) {
      float e = red[tid] + red[64 + tid] + red[128 + tid] + red[192 + tid]
              + b4[s];
      int sv = sid[(size_t)s * NATOMS + a0 + tid];
      atomicAdd(out + sv, e);   // global_atomic_add_f32
    }
  }
}

// ---------------------------------------------------------------------------
// launch
// ---------------------------------------------------------------------------
extern "C" void kernel_launch(void* const* d_in, const int* in_sizes, int n_in,
                              void* d_out, int out_size, void* d_ws,
                              size_t ws_size, hipStream_t stream) {
  const float* feat  = (const float*)d_in[0];
  const int*   sid   = (const int*)d_in[1];
  const float* W1    = (const float*)d_in[2];
  const float* b1    = (const float*)d_in[3];
  const float* W2    = (const float*)d_in[4];
  const float* b2    = (const float*)d_in[5];
  const float* W3    = (const float*)d_in[6];
  const float* b3    = (const float*)d_in[7];
  const float* W4    = (const float*)d_in[8];
  const float* b4    = (const float*)d_in[9];
  const float* shift = (const float*)d_in[10];
  float*    out = (float*)d_out;
  unsigned* ws  = (unsigned*)d_ws;   // needs WS_END*4 = 2 MB of scratch

  init_out_kernel<<<(NSTRUCT + 255) / 256, 256, 0, stream>>>(out, shift);
  prep_weights_kernel<<<(SPECIES * 512 * 32) / 256, 256, 0, stream>>>(W1, W2, W3, ws);

  dim3 grid(NATOMS / TILE_M, SPECIES);   // 512 x 4 workgroups
  size_t lds = (16384 + 8192) * sizeof(unsigned);  // 96 KB
  mlp_kernel<<<grid, 256, lds, stream>>>(feat, sid, b1, b2, b3, W4, b4, ws, out);
}
